// CavemanGPTAttentionHead_71227737637552
// MI455X (gfx1250) — matrix operand order repaired and checked by reading, not compiled, not run on hardware
//
#include <hip/hip_runtime.h>

typedef __bf16 bf16;
typedef __attribute__((ext_vector_type(4)))  __bf16 bf16x4;
typedef __attribute__((ext_vector_type(8)))  __bf16 bf16x8;
typedef __attribute__((ext_vector_type(16))) __bf16 bf16x16;
typedef __attribute__((ext_vector_type(4)))  float  f32x4;
typedef __attribute__((ext_vector_type(8)))  float  f32x8;

// ---------------------------------------------------------------- constants
#define BATCH 4
#define SEQ   2048
#define EMB   1024
#define HID   4096
#define OUTD  1024

// 16-byte global -> LDS async copy (per lane). lds_off is a byte offset inside
// the workgroup LDS allocation (dynamic-LDS base == 0 since no static LDS here).
__device__ __forceinline__ void async_copy16(unsigned lds_off, const bf16* g) {
  asm volatile("global_load_async_to_lds_b128 %0, %1, off"
               :: "v"(lds_off), "v"(g)
               : "memory");
}

// ---------------------------------------------------------------- fp32 -> bf16
__global__ __launch_bounds__(256) void cvt_f32_to_bf16_v4(
    const float* __restrict__ src, bf16* __restrict__ dst, long n4) {
  long i = (long)blockIdx.x * blockDim.x + threadIdx.x;
  long stride = (long)gridDim.x * blockDim.x;
  for (; i < n4; i += stride) {
    f32x4 v = *(const f32x4*)(src + i * 4);
    bf16x4 o;
    o[0] = (bf16)v[0]; o[1] = (bf16)v[1];
    o[2] = (bf16)v[2]; o[3] = (bf16)v[3];
    *(bf16x4*)(dst + i * 4) = o;
  }
}

// ---------------------------------------------------------------- NT GEMM via WMMA
// D[m,n] = scale * sum_k A[m,k] * Bm[n,k]   (A: MxK row-major, Bm: NxK row-major)
// TRANS=true stores D transposed: D[n*ldd + m] (used to produce V^T).
// Workgroup = 256 threads = 8 waves, tile 128(M) x 128(N); wave tile 32x64.
// A/B tiles (128x32 bf16 each) are staged in LDS via the async DMA path
// (global_load_async_to_lds_b128, ASYNCcnt) with double buffering; fragments
// are read back with ds_load_b128.  Dynamic LDS: 2 x (8KB A + 8KB B) = 32 KB.
template <typename DT, bool TRANS>
__global__ __launch_bounds__(256) void gemm_nt_wmma(
    const bf16* __restrict__ A, long saz, int lda,
    const bf16* __restrict__ Bm, long sbz, int ldb,
    DT* __restrict__ D, long sdz, int ldd,
    int K, float scale) {
  extern __shared__ __align__(16) char smem[];  // 32 KB, dynamic only (base off 0)

  const int tid  = threadIdx.x;
  const int lane = tid & 31;
  const int wave = tid >> 5;
  const int wm = wave & 3;   // 4 row-groups of 32
  const int wn = wave >> 2;  // 2 col-groups of 64
  const int lh  = lane & 15;
  const int sel = lane >> 4;

  // block bases
  A  += (long)blockIdx.z * saz + (long)blockIdx.y * 128 * lda;
  Bm += (long)blockIdx.z * sbz + (long)blockIdx.x * 128 * ldb;
  D  += (long)blockIdx.z * sdz;

  f32x8 acc[2][4];
#pragma unroll
  for (int i = 0; i < 2; ++i)
#pragma unroll
    for (int j = 0; j < 4; ++j)
#pragma unroll
      for (int v = 0; v < 8; ++v) acc[i][j][v] = 0.0f;

  // ---- async tile prefetch: 128x32 bf16 = 512 x 16B chunks per matrix;
  //      256 threads issue 2 A-chunks + 2 B-chunks each (4 async ops). ----
  auto prefetch = [&](int buf, int k0) {
    const unsigned base = (unsigned)buf * 16384u;
#pragma unroll
    for (int i = 0; i < 2; ++i) {
      const int chunk = i * 256 + tid;     // 0..511
      const int row = chunk >> 2;          // 0..127
      const int c4  = chunk & 3;           // 16B chunk within 64B row
      async_copy16(base + row * 64 + c4 * 16,
                   A + (long)row * lda + k0 + c4 * 8);
      async_copy16(base + 8192u + row * 64 + c4 * 16,
                   Bm + (long)row * ldb + k0 + c4 * 8);
    }
  };

  // ---- one 32-deep k-step from LDS buffer `buf` ----
  auto compute = [&](int buf) {
    const char* sA = smem + buf * 16384;
    const char* sB = sA + 8192;
    // A frag 16x32: lane<16 -> K 0..7 & 16..23 ; lane>=16 -> K 8..15 & 24..31
    bf16x16 af[2];
#pragma unroll
    for (int im = 0; im < 2; ++im) {
      const char* p = sA + (wm * 32 + im * 16 + lh) * 64 + sel * 16;
      bf16x8 lo = *(const bf16x8*)p;
      bf16x8 hi = *(const bf16x8*)(p + 32);
      af[im] = __builtin_shufflevector(lo, hi, 0, 1, 2, 3, 4, 5, 6, 7,
                                       8, 9, 10, 11, 12, 13, 14, 15);
    }
    // B frag 32x16: lane<16 -> K 0..7 & 16..23 ; lane>=16 -> K 8..15 & 24..31
    bf16x16 bfr[4];
#pragma unroll
    for (int jn = 0; jn < 4; ++jn) {
      const char* p = sB + (wn * 64 + jn * 16 + lh) * 64 + sel * 16;
      bf16x8 lo = *(const bf16x8*)p;
      bf16x8 hi = *(const bf16x8*)(p + 32);
      bfr[jn] = __builtin_shufflevector(lo, hi, 0, 1, 2, 3, 4, 5, 6, 7,
                                        8, 9, 10, 11, 12, 13, 14, 15);
    }
#pragma unroll
    for (int im = 0; im < 2; ++im)
#pragma unroll
      for (int jn = 0; jn < 4; ++jn)
        acc[im][jn] = __builtin_amdgcn_wmma_f32_16x16x32_bf16(
            false, af[im], false, bfr[jn], (short)0, acc[im][jn], false, false);
  };

  // ---- double-buffered main loop, last k-tile peeled (branch-free body) ----
  const int NK = K >> 5;  // always >= 32 here
  prefetch(0, 0);
  for (int kt = 0; kt < NK - 1; ++kt) {
    prefetch((kt + 1) & 1, (kt + 1) << 5);
    // in-order completion: <=4 outstanding => current buffer's 4 ops landed
    asm volatile("s_wait_asynccnt 0x4" ::: "memory");
    __syncthreads();           // whole tile visible to all waves
    compute(kt & 1);
    __syncthreads();           // all reads done before buffer is overwritten
  }
  asm volatile("s_wait_asynccnt 0x0" ::: "memory");
  __syncthreads();
  compute((NK - 1) & 1);

  // C/D layout: lane<16: VGPR v -> (M=v, N=lane); lane>=16: (M=v+8, N=lane-16)
  const int m0 = blockIdx.y * 128 + wm * 32;
  const int n0 = blockIdx.x * 128 + wn * 64;
#pragma unroll
  for (int im = 0; im < 2; ++im) {
#pragma unroll
    for (int jn = 0; jn < 4; ++jn) {
      const int col   = n0 + jn * 16 + lh;
      const int rbase = m0 + im * 16 + 8 * sel;
      if constexpr (TRANS) {
        bf16x8 pk;
#pragma unroll
        for (int v = 0; v < 8; ++v) pk[v] = (bf16)(acc[im][jn][v] * scale);
        *(bf16x8*)(D + (long)col * ldd + rbase) = pk;  // 8 consecutive M values
      } else {
#pragma unroll
        for (int v = 0; v < 8; ++v)
          D[(long)(rbase + v) * ldd + col] = (DT)(acc[im][jn][v] * scale);
      }
    }
  }
}

// ---------------------------------------------------------------- softmax (row of 2048)
// Reads fp32 row, writes bf16 probabilities in place at the row base
// (bf16 row stride stays 2*SEQ elements => no aliasing with other rows).
__global__ __launch_bounds__(256) void softmax_row_inplace_bf16(
    float* __restrict__ scores) {
  __shared__ float row[SEQ];
  __shared__ float red[256];
  const int t = threadIdx.x;
  float* rp = scores + (long)blockIdx.x * SEQ;

  float lmax = -3.402823466e38f;
#pragma unroll
  for (int i = 0; i < SEQ / 256; ++i) {
    float v = rp[t + i * 256];
    row[t + i * 256] = v;
    lmax = fmaxf(lmax, v);
  }
  red[t] = lmax;
  __syncthreads();
  for (int s = 128; s > 0; s >>= 1) {
    if (t < s) red[t] = fmaxf(red[t], red[t + s]);
    __syncthreads();
  }
  const float m = red[0];
  __syncthreads();

  float lsum = 0.0f;
#pragma unroll
  for (int i = 0; i < SEQ / 256; ++i) {
    const int j = t + i * 256;
    float e = __expf(row[j] - m);
    row[j] = e;
    lsum += e;
  }
  red[t] = lsum;
  __syncthreads();
  for (int s = 128; s > 0; s >>= 1) {
    if (t < s) red[t] += red[t + s];
    __syncthreads();
  }
  const float inv = 1.0f / red[0];

  bf16* op = (bf16*)rp;
  const int j0 = t * 8;  // 2048/256 = 8 contiguous elements per thread
  bf16x8 pk;
#pragma unroll
  for (int v = 0; v < 8; ++v) pk[v] = (bf16)(row[j0 + v] * inv);
  *(bf16x8*)(op + j0) = pk;
}

// ---------------------------------------------------------------- launch
extern "C" void kernel_launch(void* const* d_in, const int* in_sizes, int n_in,
                              void* d_out, int out_size, void* d_ws, size_t ws_size,
                              hipStream_t stream) {
  const float* emb = (const float*)d_in[0];  // [4,2048,1024]
  const float* Wq  = (const float*)d_in[1];  // [4096,1024]
  const float* Wk  = (const float*)d_in[2];  // [4096,1024]
  const float* Wv  = (const float*)d_in[3];  // [1024,1024]
  float* out = (float*)d_out;                // [4,2048,1024] fp32

  // ---- workspace layout (bytes) ----
  char* ws = (char*)d_ws;
  size_t off = 0;
  bf16* embB = (bf16*)(ws + off); off += (size_t)BATCH * SEQ * EMB * 2;   // 16 MB
  bf16* WqB  = (bf16*)(ws + off); off += (size_t)HID * EMB * 2;          //  8 MB
  bf16* WkB  = (bf16*)(ws + off); off += (size_t)HID * EMB * 2;          //  8 MB
  bf16* WvB  = (bf16*)(ws + off); off += (size_t)OUTD * EMB * 2;         //  2 MB
  bf16* Qb   = (bf16*)(ws + off); off += (size_t)BATCH * SEQ * HID * 2;  // 64 MB
  bf16* Kb   = (bf16*)(ws + off); off += (size_t)BATCH * SEQ * HID * 2;  // 64 MB
  bf16* Vtb  = (bf16*)(ws + off); off += (size_t)BATCH * OUTD * SEQ * 2; // 16 MB
  float* Sc  = (float*)(ws + off);                                       // 64 MB

  const size_t LDS_BYTES = 32768;  // 2 x (8KB A + 8KB B)

  // ---- 1) fp32 -> bf16 conversions ----
  {
    long n;
    n = (long)BATCH * SEQ * EMB / 4;
    cvt_f32_to_bf16_v4<<<2048, 256, 0, stream>>>(emb, embB, n);
    n = (long)HID * EMB / 4;
    cvt_f32_to_bf16_v4<<<2048, 256, 0, stream>>>(Wq, WqB, n);
    cvt_f32_to_bf16_v4<<<2048, 256, 0, stream>>>(Wk, WkB, n);
    n = (long)OUTD * EMB / 4;
    cvt_f32_to_bf16_v4<<<1024, 256, 0, stream>>>(Wv, WvB, n);
  }

  // ---- 2) projections ----
  // Q[b,s,h] = emb[b,s,:] . Wq[h,:]  (scaled by 1/sqrt(HID)=1/64, folded here)
  gemm_nt_wmma<bf16, false>
      <<<dim3(HID / 128, SEQ / 128, BATCH), 256, LDS_BYTES, stream>>>(
      embB, (long)SEQ * EMB, EMB, WqB, 0, EMB,
      Qb, (long)SEQ * HID, HID, EMB, 1.0f / 64.0f);
  // K[b,s,h]
  gemm_nt_wmma<bf16, false>
      <<<dim3(HID / 128, SEQ / 128, BATCH), 256, LDS_BYTES, stream>>>(
      embB, (long)SEQ * EMB, EMB, WkB, 0, EMB,
      Kb, (long)SEQ * HID, HID, EMB, 1.0f);
  // V^T[b,o,s] = emb[b,s,:] . Wv[o,:]   (transposed store)
  gemm_nt_wmma<bf16, true>
      <<<dim3(OUTD / 128, SEQ / 128, BATCH), 256, LDS_BYTES, stream>>>(
      embB, (long)SEQ * EMB, EMB, WvB, 0, EMB,
      Vtb, (long)OUTD * SEQ, SEQ, EMB, 1.0f);

  // ---- 3) scores[b,i,j] = K[b,i,:] . Q[b,j,:]  (already / sqrt(d_k)) ----
  gemm_nt_wmma<float, false>
      <<<dim3(SEQ / 128, SEQ / 128, BATCH), 256, LDS_BYTES, stream>>>(
      Kb, (long)SEQ * HID, HID, Qb, (long)SEQ * HID, HID,
      Sc, (long)SEQ * SEQ, SEQ, HID, 1.0f);

  // ---- 4) softmax rows, bf16 probabilities written in place ----
  softmax_row_inplace_bf16<<<BATCH * SEQ, 256, 0, stream>>>(Sc);

  // ---- 5) out[b,i,o] = sum_j attn[b,i,j] * V[b,j,o] = attn . (V^T rows) ----
  // attn is bf16 at Sc base with row stride 2*SEQ elements.
  gemm_nt_wmma<float, false>
      <<<dim3(OUTD / 128, SEQ / 128, BATCH), 256, LDS_BYTES, stream>>>(
      (const bf16*)Sc, (long)SEQ * 2 * SEQ, 2 * SEQ,
      Vtb, (long)OUTD * SEQ, SEQ,
      out, (long)SEQ * OUTD, OUTD, SEQ, 1.0f);
}
